// CrossAttention_86586540687585
// MI455X (gfx1250) — compile-verified, matrix operand order checked by reading, not run on hardware
//
#include <hip/hip_runtime.h>

// ---------------------------------------------------------------------------
// Problem constants
// ---------------------------------------------------------------------------
#define BB    256
#define TT    200
#define NH    8
#define DD    64
#define MODEL 512        // NH*DD
#define FFN   2048
#define NTOK  (BB*TT)    // 51200
#define EPS   1.1920929e-07f

// FFN tiling
#define RM 64            // tokens per block
#define KC 128           // FFN_H chunk per outer iteration
#define PX 520           // LDS pitch (bf16 elems) for X tile (512 + 4-dword pad)
#define PG 136           // LDS pitch (bf16 elems) for G chunk

typedef __attribute__((ext_vector_type(16))) __bf16 v16bf;
typedef __attribute__((ext_vector_type(8)))  float  v8f;
typedef __attribute__((ext_vector_type(4)))  unsigned int u32x4;
typedef __attribute__((ext_vector_type(8)))  int i32x8;
typedef __attribute__((ext_vector_type(4)))  int i32x4;

#if defined(__AMDGCN__) && __has_builtin(__builtin_amdgcn_tensor_load_to_lds) && \
    __has_builtin(__builtin_amdgcn_s_wait_tensorcnt)
#define HAVE_TDM 1
#else
#define HAVE_TDM 0
#endif

union Frag {             // one WMMA 16x32 bf16 operand (8 VGPRs / lane)
    uint4 q[2];
    v16bf v;
};

__device__ __forceinline__ unsigned short f2bf(float x) {
    unsigned int u = __float_as_uint(x);
    unsigned int r = u + 0x7FFFu + ((u >> 16) & 1u);   // round-to-nearest-even
    return (unsigned short)(r >> 16);
}
__device__ __forceinline__ float bf2f(unsigned short h) {
    return __uint_as_float(((unsigned int)h) << 16);
}

// ---------------------------------------------------------------------------
// Kernel 1: convert + transpose weights to bf16, K-contiguous per output col
//   W1bT[n][k] = w1[k][n]   (2048 x 512)
//   W2bT[n][k] = w2[k][n]   (2048 x 512)
//   W3bT[n][k] = w3[k][n]   (512 x 2048)
// ---------------------------------------------------------------------------
__global__ void prep_weights(const float* __restrict__ w1,
                             const float* __restrict__ w2,
                             const float* __restrict__ w3,
                             unsigned short* __restrict__ W1bT,
                             unsigned short* __restrict__ W2bT,
                             unsigned short* __restrict__ W3bT) {
    int idx = blockIdx.x * 256 + threadIdx.x;
    if (idx < 1048576) {
        int n = idx >> 9, k = idx & 511;
        W1bT[idx] = f2bf(w1[(size_t)k * FFN + n]);
    } else if (idx < 2097152) {
        int t = idx - 1048576;
        int n = t >> 9, k = t & 511;
        W2bT[t] = f2bf(w2[(size_t)k * FFN + n]);
    } else if (idx < 3145728) {
        int t = idx - 2097152;
        int n = t >> 11, k = t & 2047;
        W3bT[t] = f2bf(w3[(size_t)k * MODEL + n]);
    }
}

// ---------------------------------------------------------------------------
// Kernel 2: RMSNorm -> bf16 X.  One wave per token, 8 tokens per block.
// ---------------------------------------------------------------------------
__global__ void rmsnorm_kernel(const float* __restrict__ seq,
                               const float* __restrict__ rms_w,
                               unsigned short* __restrict__ Xb) {
    const int wid  = threadIdx.x >> 5;
    const int lane = threadIdx.x & 31;
    const int g    = blockIdx.x * 8 + wid;            // token id
    const float* row = seq + (size_t)g * MODEL;
    float x[16];
    float ss = 0.f;
#pragma unroll
    for (int j = 0; j < 16; ++j) {
        x[j] = row[lane + 32 * j];
        ss += x[j] * x[j];
    }
#pragma unroll
    for (int m = 16; m >= 1; m >>= 1) ss += __shfl_xor(ss, m, 32);
    const float scale = rsqrtf(ss * (1.f / (float)MODEL) + EPS);
#pragma unroll
    for (int j = 0; j < 16; ++j) {
        int c = lane + 32 * j;
        Xb[(size_t)g * MODEL + c] = f2bf(x[j] * scale * rms_w[c]);
    }
}

// ---------------------------------------------------------------------------
// Kernel 3: fused SwiGLU FFN + residual, bf16 WMMA, fp32 accumulate.
//   H = silu(X@W1) * (X@W2) @ W3 + seq      (per 64-token tile)
// 8 waves / block.  X tile staged into LDS by the Tensor Data Mover (TDM)
// with hardware row padding (512 -> 520 bf16 pitch).  Stage 1 writes a 64xKC
// bf16 G chunk to LDS; stage 2 accumulates H(64x512) in registers.
// ---------------------------------------------------------------------------
__global__ __launch_bounds__(256, 1)
void ffn_kernel(const unsigned short* __restrict__ Xb,
                const unsigned short* __restrict__ W1bT,
                const unsigned short* __restrict__ W2bT,
                const unsigned short* __restrict__ W3bT,
                const float* __restrict__ seq,
                unsigned short* __restrict__ Hb) {
    __shared__ unsigned short lds_X[RM * PX];   // 66,560 B
    __shared__ unsigned short lds_G[RM * PG];   // 17,408 B

    const int tid  = threadIdx.x;
    const int lane = tid & 31;
    const int wid  = tid >> 5;
    const int m0   = blockIdx.x * RM;
    const int hf   = lane >> 4;       // lane half (0/1)
    const int lr   = lane & 15;

    // ---- stage the 64x512 bf16 X tile into LDS (row pitch PX) ----
#if HAVE_TDM
    if (wid == 0) {
        // Tensor DMA descriptor (ISA 8.3/8.4): 2D tile 512x64 bf16,
        // LDS padding: 4 dwords (8 bf16) after every 256 dwords (512 bf16).
        const unsigned long long gaddr =
            (unsigned long long)(const void*)(Xb + (size_t)m0 * MODEL);
        const unsigned int laddr =
            (unsigned int)(unsigned long long)(const void*)&lds_X[0];
        u32x4 g0;
        g0[0] = 1u;                                       // count=1 (valid D#)
        g0[1] = laddr;                                    // lds_addr
        g0[2] = (unsigned int)gaddr;                      // global_addr[31:0]
        g0[3] = (unsigned int)((gaddr >> 32) & 0x1FFFFFFu) | (2u << 30); // type=2
        i32x8 g1;
        g1[0] = (int)((1u << 16)      // data_size = 2 bytes
                    | (1u << 20)      // pad_enable
                    | (7u << 22)      // pad_interval: every 256 dwords
                    | (3u << 25));    // pad_amount:   4 dwords
        g1[1] = (int)(512u << 16);    // tensor_dim0[15:0] in [63:48]
        g1[2] = (int)(64u << 16);     // tensor_dim0 hi=0, tensor_dim1 lo=64
        g1[3] = (int)(512u << 16);    // tensor_dim1 hi=0, tile_dim0=512
        g1[4] = 64;                   // tile_dim1=64, tile_dim2=0
        g1[5] = 512;                  // tensor_dim0_stride[31:0]
        g1[6] = 0;                    // stride0 hi, tensor_dim1_stride lo
        g1[7] = 0;
        i32x4 gz = {0, 0, 0, 0};
#if __clang_major__ >= 23
        i32x8 gz8 = {0, 0, 0, 0, 0, 0, 0, 0};
        __builtin_amdgcn_tensor_load_to_lds(g0, g1, gz, gz, gz8, 0);
#else
        __builtin_amdgcn_tensor_load_to_lds(g0, g1, gz, gz, 0);
#endif
        __builtin_amdgcn_s_wait_tensorcnt(0);
    }
#else
    {
        const uint4* src = (const uint4*)(Xb + (size_t)m0 * MODEL);
        for (int idx = tid; idx < RM * (MODEL / 8); idx += 256) {
            int r = idx >> 6, c = idx & 63;              // 64 uint4 per row
            uint4 d = src[r * 64 + c];
            *(uint4*)&lds_X[r * PX + c * 8] = d;
        }
    }
#endif
    __syncthreads();

    // stage-2 accumulators: wave (mt2, ntsel) owns 16 of the 128 H tiles
    const int mt1 = wid & 3, ntg = wid >> 2;     // stage-1 assignment
    const int mt2 = wid >> 1, ntsel = wid & 1;   // stage-2 assignment

    v8f acc2[16];
#pragma unroll
    for (int j = 0; j < 16; ++j) acc2[j] = {};

    for (int cc = 0; cc < FFN / KC; ++cc) {
        // prefetch next chunk's gate/up weights while this chunk computes
        if (cc + 1 < FFN / KC) {
            size_t nb = (size_t)(cc + 1) * KC * MODEL +
                        (size_t)(wid * 16 + (lane >> 1)) * MODEL;
            __builtin_prefetch(W1bT + nb, 0, 1);
            __builtin_prefetch(W2bT + nb, 0, 1);
        }
        // ---------------- stage 1: G = silu(X@W1) * (X@W2), KC cols -------
        const int arow = mt1 * 16 + lr;
        for (int jj = 0; jj < 2; ++jj) {
            v8f accA[2], accB[2];
#pragma unroll
            for (int j = 0; j < 2; ++j) { accA[j] = {}; accB[j] = {}; }
            for (int ks = 0; ks < MODEL / 32; ++ks) {
                Frag a;
                const unsigned short* ap = &lds_X[arow * PX + ks * 32 + hf * 8];
                a.q[0] = *(const uint4*)ap;
                a.q[1] = *(const uint4*)(ap + 16);
#pragma unroll
                for (int j = 0; j < 2; ++j) {
                    const int n = cc * KC + (ntg * 4 + jj * 2 + j) * 16 + lr;
                    const unsigned short* bp1 =
                        W1bT + (size_t)n * MODEL + ks * 32 + hf * 8;
                    const unsigned short* bp2 =
                        W2bT + (size_t)n * MODEL + ks * 32 + hf * 8;
                    Frag b1, b2;
                    b1.q[0] = *(const uint4*)bp1;
                    b1.q[1] = *(const uint4*)(bp1 + 16);
                    b2.q[0] = *(const uint4*)bp2;
                    b2.q[1] = *(const uint4*)(bp2 + 16);
                    accA[j] = __builtin_amdgcn_wmma_f32_16x16x32_bf16(
                        false, a.v, false, b1.v, (short)0, accA[j], false, false);
                    accB[j] = __builtin_amdgcn_wmma_f32_16x16x32_bf16(
                        false, a.v, false, b2.v, (short)0, accB[j], false, false);
                }
            }
            // silu(a)*b -> bf16 -> LDS G chunk
#pragma unroll
            for (int j = 0; j < 2; ++j) {
                const int col = (ntg * 4 + jj * 2 + j) * 16 + lr;
#pragma unroll
                for (int i = 0; i < 8; ++i) {
                    const int row = mt1 * 16 + hf * 8 + i;
                    float av = accA[j][i], bv = accB[j][i];
                    float g = (av / (1.f + __expf(-av))) * bv;
                    lds_G[row * PG + col] = f2bf(g);
                }
            }
        }
        __syncthreads();

        // ---------------- stage 2: H += G_chunk @ W3[cc*KC : , :] ---------
        const int arow2 = mt2 * 16 + lr;
        for (int ks = 0; ks < KC / 32; ++ks) {
            Frag g;
            const unsigned short* gp = &lds_G[arow2 * PG + ks * 32 + hf * 8];
            g.q[0] = *(const uint4*)gp;
            g.q[1] = *(const uint4*)(gp + 16);
#pragma unroll
            for (int j = 0; j < 16; ++j) {
                const int n = (j * 2 + ntsel) * 16 + lr;
                const unsigned short* bp =
                    W3bT + (size_t)n * FFN + cc * KC + ks * 32 + hf * 8;
                Frag b3;
                b3.q[0] = *(const uint4*)bp;
                b3.q[1] = *(const uint4*)(bp + 16);
                acc2[j] = __builtin_amdgcn_wmma_f32_16x16x32_bf16(
                    false, g.v, false, b3.v, (short)0, acc2[j], false, false);
            }
        }
        __syncthreads();   // lds_G reused next chunk
    }

    // ---- epilogue: + residual, store bf16 H ----
#pragma unroll
    for (int j = 0; j < 16; ++j) {
        const int col = (j * 2 + ntsel) * 16 + lr;
#pragma unroll
        for (int i = 0; i < 8; ++i) {
            const int row = mt2 * 16 + hf * 8 + i;
            const size_t idx = (size_t)(m0 + row) * MODEL + col;
            float h = acc2[j][i] + seq[idx];
            Hb[idx] = f2bf(h);
        }
    }
}

// ---------------------------------------------------------------------------
// Kernel 4: collapsed single-query attention + residual (fp32, tiny).
//   r = w_k[n]@q ; scores = h·r ; softmax ; u = attnᵀ·h ; ctx = u@w_v[n] + q
// One 64-thread block per (b, head).
// ---------------------------------------------------------------------------
__global__ void attn_kernel(const float* __restrict__ q,
                            const unsigned short* __restrict__ Hb,
                            const float* __restrict__ w_k,
                            const float* __restrict__ w_v,
                            const unsigned char* __restrict__ mask,
                            float* __restrict__ out) {
    const int n = blockIdx.x & 7;
    const int b = blockIdx.x >> 3;
    const int tid = threadIdx.x;                 // 0..63

    __shared__ float s_q[64], s_r[64], s_u[64], s_attn[TT], s_red[64];

    s_q[tid] = q[(size_t)(b * NH + n) * DD + tid];
    __syncthreads();

    // r = w_k[n] @ q
    {
        const float* wk = w_k + (size_t)n * DD * DD + (size_t)tid * DD;
        float s = 0.f;
        for (int e = 0; e < DD; ++e) s += wk[e] * s_q[e];
        s_r[tid] = s;
    }
    __syncthreads();

    // scores
    const float scale = 0.125f;                  // 1/sqrt(64)
    float mloc = -1e30f;
    for (int t = tid; t < TT; t += 64) {
        const unsigned short* hp = Hb + ((size_t)(b * TT + t) * MODEL + n * DD);
        float s = 0.f;
        for (int d = 0; d < DD; ++d) s += bf2f(hp[d]) * s_r[d];
        s *= scale;
        if (!mask[b * TT + t]) s = -1e30f;
        s_attn[t] = s;
        mloc = fmaxf(mloc, s);
    }
    s_red[tid] = mloc;
    for (int off = 32; off >= 1; off >>= 1) {
        __syncthreads();
        if (tid < off) s_red[tid] = fmaxf(s_red[tid], s_red[tid + off]);
    }
    __syncthreads();
    const float mx = s_red[0];
    __syncthreads();

    // exp + sum
    float sloc = 0.f;
    for (int t = tid; t < TT; t += 64) {
        float e = __expf(s_attn[t] - mx);
        s_attn[t] = e;
        sloc += e;
    }
    s_red[tid] = sloc;
    for (int off = 32; off >= 1; off >>= 1) {
        __syncthreads();
        if (tid < off) s_red[tid] += s_red[tid + off];
    }
    __syncthreads();
    const float tot = s_red[0];

    // u[d] = sum_t attn[t] * h[t,d]
    float u = 0.f;
    for (int t = 0; t < TT; ++t)
        u += s_attn[t] * bf2f(Hb[(size_t)(b * TT + t) * MODEL + n * DD + tid]);
    s_u[tid] = u;
    __syncthreads();

    // ctx = (u/tot) @ w_v[n]  + q
    float c = 0.f;
    for (int d = 0; d < DD; ++d)
        c += s_u[d] * w_v[(size_t)n * DD * DD + (size_t)d * DD + tid];
    out[(size_t)(b * NH + n) * DD + tid] = c / tot + s_q[tid];
}

// ---------------------------------------------------------------------------
// Launch
// ---------------------------------------------------------------------------
extern "C" void kernel_launch(void* const* d_in, const int* in_sizes, int n_in,
                              void* d_out, int out_size, void* d_ws, size_t ws_size,
                              hipStream_t stream) {
    const float* q     = (const float*)d_in[0];
    const float* seq   = (const float*)d_in[1];
    const float* rms_w = (const float*)d_in[2];
    const float* w1    = (const float*)d_in[3];
    const float* w2    = (const float*)d_in[4];
    const float* w3    = (const float*)d_in[5];
    const float* w_k   = (const float*)d_in[6];
    const float* w_v   = (const float*)d_in[7];
    const unsigned char* mask = (const unsigned char*)d_in[8];
    float* out = (float*)d_out;

    char* ws = (char*)d_ws;
    // scratch layout (bytes)
    unsigned short* Xb   = (unsigned short*)(ws);                 // 52,428,800
    unsigned short* Hb   = (unsigned short*)(ws + 52428800);      // 52,428,800
    unsigned short* W1bT = (unsigned short*)(ws + 104857600);     //  2,097,152
    unsigned short* W2bT = (unsigned short*)(ws + 106954752);     //  2,097,152
    unsigned short* W3bT = (unsigned short*)(ws + 109051904);     //  2,097,152

    prep_weights<<<12288, 256, 0, stream>>>(w1, w2, w3, W1bT, W2bT, W3bT);
    rmsnorm_kernel<<<NTOK / 8, 256, 0, stream>>>(seq, rms_w, Xb);
    ffn_kernel<<<NTOK / RM, 256, 0, stream>>>(Xb, W1bT, W2bT, W3bT, seq, Hb);
    attn_kernel<<<BB * NH, 64, 0, stream>>>(q, Hb, w_k, w_v, mask, out);
}
